// MessageFunction_28028956573795
// MI455X (gfx1250) — matrix operand order repaired and checked by reading, not compiled, not run on hardware
//
#include <hip/hip_runtime.h>
#include <hip/hip_bf16.h>
#include <stdint.h>

#define BATCH   32
#define NNODES  1024
#define DDIM    64
#define NCLS    4

typedef __attribute__((ext_vector_type(16))) __bf16 v16bf;
typedef __attribute__((ext_vector_type(8)))  float  v8f;
typedef int v4i_vs __attribute__((vector_size(16)));

#if __has_builtin(__builtin_amdgcn_global_load_async_to_lds_b128) && \
    __has_builtin(__builtin_amdgcn_s_wait_asynccnt)
#define HAVE_ASYNC_LDS 1
#define ASYNC_WAIT(n) __builtin_amdgcn_s_wait_asynccnt(n)
#else
#define HAVE_ASYNC_LDS 0
#define ASYNC_WAIT(n)
#endif

__device__ __forceinline__ unsigned short f32_to_bf16_bits(float f) {
    unsigned int x = __float_as_uint(f);
    unsigned int r = (x + 0x7FFFu + ((x >> 16) & 1u)) >> 16;  // round-nearest-even
    return (unsigned short)r;
}
__device__ __forceinline__ float bf16_bits_to_f32(unsigned short u) {
    return __uint_as_float(((unsigned int)u) << 16);
}

// 16-byte global->LDS copy: async direct-to-LDS path (ASYNCcnt tracked),
// synchronous VGPR round-trip fallback otherwise.
__device__ __forceinline__ void copy16_to_lds(const void* gsrc, void* ldst) {
#if HAVE_ASYNC_LDS
    __builtin_amdgcn_global_load_async_to_lds_b128(
        (__attribute__((address_space(1))) v4i_vs*)gsrc,
        (__attribute__((address_space(3))) v4i_vs*)ldst, 0, 0);
#else
    *(uint4*)ldst = *(const uint4*)gsrc;
#endif
}

// ---------------------------------------------------------------------------
// Kernel 1: per (b, j): acc_c = W[dir,c] @ h[b,j]  (f32)
//   slots c=0..2 store bf16(acc_c - acc_3)  (delta vs class 3)
//   slot  c=3   stores bf16(acc_3)
// j-contiguous layout so kernel 3's WMMA B-operand is a contiguous 32B read.
// ---------------------------------------------------------------------------
__global__ void __launch_bounds__(256)
ggnn_transform_kernel(const float* __restrict__ node_state,
                      const float* __restrict__ matrix_in,
                      const float* __restrict__ matrix_out,
                      unsigned short* __restrict__ g_in,
                      unsigned short* __restrict__ g_out) {
    __shared__ float h_lds[64][65];            // pad 65: conflict-free column reads
    const int jb  = blockIdx.x;                // 0..15 (j tile of 64)
    const int b   = blockIdx.y;                // 0..31
    const int tid = threadIdx.x;
    const int j0  = jb * 64;

    for (int u = tid; u < 64 * 64; u += 256) {
        int row = u >> 6, e = u & 63;
        h_lds[row][e] = node_state[((size_t)(b * NNODES + j0 + row)) * DDIM + e];
    }
    __syncthreads();

    // 2 dirs * 64 d * 64 jl = 8192 (dir,d,jl) combos -> 32 per thread, 4 classes each
    for (int it = 0; it < 32; ++it) {
        int idx  = tid + it * 256;
        int jl   = idx & 63;
        int rest = idx >> 6;                   // 0..127
        int d    = rest & 63;
        int dir  = rest >> 6;
        const float* mat = dir ? matrix_out : matrix_in;
        float acc[NCLS] = {0.f, 0.f, 0.f, 0.f};
        #pragma unroll
        for (int e = 0; e < DDIM; ++e) {
            float hv = h_lds[jl][e];
            #pragma unroll
            for (int c = 0; c < NCLS; ++c)
                acc[c] += mat[(c * DDIM + d) * DDIM + e] * hv;
        }
        unsigned short* dst = dir ? g_out : g_in;
        const size_t pos = (size_t)(j0 + jl);
        #pragma unroll
        for (int c = 0; c < 3; ++c)
            dst[(((size_t)(b * NCLS + c)) * DDIM + d) * NNODES + pos] =
                f32_to_bf16_bits(acc[c] - acc[3]);
        dst[(((size_t)(b * NCLS + 3)) * DDIM + d) * NNODES + pos] =
            f32_to_bf16_bits(acc[3]);
    }
}

// ---------------------------------------------------------------------------
// Kernel 2: T[dir][b][d] = sum_j g3[b,j,d]   (i-independent baseline term)
// One block per (dir,b); deterministic fixed-order reduction (no atomics).
// ---------------------------------------------------------------------------
__global__ void __launch_bounds__(256)
ggnn_colsum_kernel(const unsigned short* __restrict__ g_in,
                   const unsigned short* __restrict__ g_out,
                   float* __restrict__ T) {
    const int blk = blockIdx.x;                // 0..63
    const int b   = blk & 31;
    const int dir = blk >> 5;
    const unsigned short* g = dir ? g_out : g_in;
    const int tid = threadIdx.x;
    const int d   = tid & 63;
    const int q   = tid >> 6;                  // 0..3
    const unsigned short* row =
        g + (((size_t)(b * NCLS + 3)) * DDIM + d) * NNODES + q * 256;
    float s = 0.f;
    for (int j = 0; j < 256; ++j) s += bf16_bits_to_f32(row[j]);
    __shared__ float red[4][64];
    red[q][d] = s;
    __syncthreads();
    if (q == 0)
        T[((size_t)dir * BATCH + b) * DDIM + d] =
            (red[0][d] + red[1][d]) + (red[2][d] + red[3][d]);
}

// ---------------------------------------------------------------------------
// Kernel 3: m_dir[b,i,:] = T_dir[b,:] + sum_{c<3} sum_j onehot(adj,c)*gd_dir[b,j,c,:]
// bf16 WMMA matmuls (A = exact 0/1 one-hot).  Block = 4 waves = 64 i rows.
// Double-buffered async global->LDS staging pipelined against WMMA compute.
// ---------------------------------------------------------------------------
__global__ void __launch_bounds__(128)
ggnn_message_kernel(const int* __restrict__ adj,
                    const unsigned short* __restrict__ g_in,
                    const unsigned short* __restrict__ g_out,
                    const float* __restrict__ bias,
                    const float* __restrict__ T,
                    float* __restrict__ out) {
    __shared__ alignas(64) unsigned short gin_lds [2][3][DDIM][32];  // 24 KB
    __shared__ alignas(64) unsigned short gout_lds[2][3][DDIM][32];  // 24 KB
    __shared__ int adjin_lds [2][64][36];  // rows i0.., cols j chunk (pad 36)
    __shared__ int adjout_lds[2][32][68];  // rows j chunk, cols i0.. (pad 68)

    const int ib   = blockIdx.x;         // 0..15
    const int b    = blockIdx.y;         // 0..31
    const int tid  = threadIdx.x;
    const int lane = tid & 31;
    const int w    = tid >> 5;           // wave id 0..3 -> i sub-tile
    const int i0   = ib * 64;
    const int mrow = lane & 15;          // M row / N col within 16
    const int kb   = (lane & 16);        // 0 or 16 (A-layout K base)
    const int hi   = (lane & 16) >> 1;   // 0 or 8 (C/D row offset)

    // 20 async 16B transfers per thread per chunk (12 g + 8 adj)
    auto stage = [&](int ch) {
        const int j0 = ch * 32;
        const int p  = ch & 1;
        #pragma unroll
        for (int k = 0; k < 6; ++k) {
            int u  = tid + k * 128;          // 0..767
            int cd = u >> 2, seg = u & 3;    // (c*64+d) with c<3, 16B segment
            copy16_to_lds((const uint4*)(g_in  + ((size_t)(b * 256 + cd)) * NNODES + j0) + seg,
                          (uint4*)(&gin_lds[p][0][0][0]) + u);
            copy16_to_lds((const uint4*)(g_out + ((size_t)(b * 256 + cd)) * NNODES + j0) + seg,
                          (uint4*)(&gout_lds[p][0][0][0]) + u);
        }
        #pragma unroll
        for (int k = 0; k < 4; ++k) {
            int u = tid + k * 128;           // 0..511
            {   // in-dir: adj[b, i0+r, j0..j0+31]
                int r = u >> 3, seg = u & 7;
                copy16_to_lds((const uint4*)(adj + ((size_t)b * NNODES + i0 + r) * NNODES + j0) + seg,
                              (uint4*)(&adjin_lds[p][r][seg * 4]));
            }
            {   // out-dir: adj[b, j0+r2, i0..i0+63]
                int r2 = u >> 4, seg2 = u & 15;
                copy16_to_lds((const uint4*)(adj + ((size_t)b * NNODES + j0 + r2) * NNODES + i0) + seg2,
                              (uint4*)(&adjout_lds[p][r2][seg2 * 4]));
            }
        }
    };

    v8f accin[4]  = {};
    v8f accout[4] = {};

    stage(0);
    for (int ch = 0; ch < 32; ++ch) {
        const int p = ch & 1;
        __syncthreads();                 // everyone done with buffer (ch+1)&1
        if (ch + 1 < 32) {
            stage(ch + 1);               // overlap next chunk with this compute
            if (ch + 2 < 32)             // speculative prefetch 2 chunks ahead
                __builtin_prefetch(adj + ((size_t)b * NNODES + i0 + (tid >> 1)) * NNODES
                                       + (ch + 2) * 32, 0, 0);
            ASYNC_WAIT(20);              // in-order: drains chunk ch, keeps ch+1 in flight
        } else {
            ASYNC_WAIT(0);               // last chunk: drain everything
        }
        __syncthreads();                 // chunk ch visible to all waves

        // ---- per-lane adjacency values in A-operand order ----
        // k order used here (contiguous, same as B): lanes<16: K = 0..15; lanes>=16: K = 16..31
        const int* rowp = &adjin_lds[p][w * 16 + mrow][kb];
        int4 q0 = *(const int4*)(rowp);
        int4 q1 = *(const int4*)(rowp + 4);
        int4 q2 = *(const int4*)(rowp + 8);
        int4 q3 = *(const int4*)(rowp + 12);
        int vin[16] = {q0.x,q0.y,q0.z,q0.w, q1.x,q1.y,q1.z,q1.w,
                       q2.x,q2.y,q2.z,q2.w, q3.x,q3.y,q3.z,q3.w};
        int vout[16];
        #pragma unroll
        for (int h = 0; h < 16; ++h) {
            int K = h + kb;              // same A layout, transposed source
            vout[h] = adjout_lds[p][K][w * 16 + mrow];
        }

        const int jl = (lane & 16);      // B-operand: lanes>=16 hold K=16..31
        #pragma unroll
        for (int c = 0; c < 3; ++c) {    // class 3 folded into T baseline
            union { v16bf v; unsigned short s[16]; } ain, aout;
            #pragma unroll
            for (int h = 0; h < 16; ++h) ain.s[h]  = (vin[h]  == c) ? 0x3F80 : 0;
            #pragma unroll
            for (int h = 0; h < 16; ++h) aout.s[h] = (vout[h] == c) ? 0x3F80 : 0;
            #pragma unroll
            for (int nt = 0; nt < 4; ++nt) {
                int d = nt * 16 + mrow;
                const v16bf* bin  = (const v16bf*)(&gin_lds [p][c][d][jl]);
                const v16bf* bout = (const v16bf*)(&gout_lds[p][c][d][jl]);
                accin[nt]  = __builtin_amdgcn_wmma_f32_16x16x32_bf16(
                                 false, ain.v,  false, *bin,  (short)0, accin[nt],  false, false);
                accout[nt] = __builtin_amdgcn_wmma_f32_16x16x32_bf16(
                                 false, aout.v, false, *bout, (short)0, accout[nt], false, false);
            }
        }
    }

    // ---- epilogue: out[b,i,0:64]=m_in+bias, [64:128]=m_out+bias (C/D layout) ----
    const size_t orow = (size_t)b * NNODES + i0 + w * 16;
    #pragma unroll
    for (int nt = 0; nt < 4; ++nt) {
        int d = nt * 16 + mrow;
        float bin  = bias[d]      + T[(size_t)b * DDIM + d];
        float bout = bias[64 + d] + T[((size_t)BATCH + b) * DDIM + d];
        #pragma unroll
        for (int r = 0; r < 8; ++r) {
            int M = r + hi;
            out[(orow + M) * 128 + d]      = accin[nt][r]  + bin;
            out[(orow + M) * 128 + 64 + d] = accout[nt][r] + bout;
        }
    }
}

extern "C" void kernel_launch(void* const* d_in, const int* in_sizes, int n_in,
                              void* d_out, int out_size, void* d_ws, size_t ws_size,
                              hipStream_t stream) {
    const float* node_state = (const float*)d_in[0];
    const int*   adj        = (const int*)d_in[1];
    const float* matrix_in  = (const float*)d_in[2];
    const float* matrix_out = (const float*)d_in[3];
    const float* bias       = (const float*)d_in[4];
    float* out = (float*)d_out;

    const size_t gelems = (size_t)BATCH * NCLS * DDIM * NNODES;   // 8.39M bf16
    unsigned short* g_in  = (unsigned short*)d_ws;                 // 16 MB
    unsigned short* g_out = g_in + gelems;                         // +16 MB
    float*          T     = (float*)(g_out + gelems);              // +16 KB

    dim3 g1(16, 32);
    ggnn_transform_kernel<<<g1, dim3(256), 0, stream>>>(node_state, matrix_in, matrix_out,
                                                        g_in, g_out);
    ggnn_colsum_kernel<<<dim3(64), dim3(256), 0, stream>>>(g_in, g_out, T);
    dim3 g2(16, 32);
    ggnn_message_kernel<<<g2, dim3(128), 0, stream>>>(adj, g_in, g_out, bias, T, out);
}
